// GATAE_74474732912749
// MI455X (gfx1250) — matrix operand order repaired and checked by reading, not compiled, not run on hardware
//
#include <hip/hip_runtime.h>
#include <hip/hip_bf16.h>

typedef __attribute__((ext_vector_type(16))) _Float16 v16h;
typedef __attribute__((ext_vector_type(8)))  float    v8f;
typedef __attribute__((ext_vector_type(4)))  unsigned u32x4;

#define N_NODES 10000
#define N_EDGES 320000
#define IN_F    256
#define HID     256
#define LATENT  128
#define NHEAD   4
#define DH1     64
#define NOUT2   512   // NHEAD*LATENT

// ---------- helpers ----------
__device__ __forceinline__ unsigned f2o(float f) {
    unsigned b = __float_as_uint(f);
    return (b & 0x80000000u) ? ~b : (b | 0x80000000u);
}
__device__ __forceinline__ float o2f(unsigned u) {
    return __uint_as_float((u & 0x80000000u) ? (u ^ 0x80000000u) : ~u);
}

// ---------- fill ----------
__global__ void k_fill_u32(unsigned* p, unsigned v, long n) {
    long i = (long)blockIdx.x * blockDim.x + threadIdx.x;
    long s = (long)gridDim.x * blockDim.x;
    for (; i < n; i += s) p[i] = v;
}

// ---------- casts ----------
__global__ void k_cast_h(const float* x, _Float16* y, long n) {
    long i = (long)blockIdx.x * blockDim.x + threadIdx.x;
    if (i < n) y[i] = (_Float16)x[i];
}
// W is [K, M] row-major; produce WT [M, K] in f16
__global__ void k_cast_hT(const float* W, _Float16* WT, int K, int M) {
    long t = (long)blockIdx.x * blockDim.x + threadIdx.x;
    if (t >= (long)K * M) return;
    int m = (int)(t / K);
    int k = (int)(t - (long)m * K);
    WT[t] = (_Float16)W[(long)k * M + m];
}

// ---------- WMMA fragment loaders ----------
// A-matrix 16x32 f16 layout: lane<16 -> K = {kk+0..7, kk+16..23}; lane>=16 -> +8
__device__ __forceinline__ v16h load_fragA(const _Float16* base, int row, int ldk,
                                           int kk, int lane) {
    int kb = (lane < 16) ? 0 : 8;
    const _Float16* p = base + (size_t)row * ldk + kk + kb;
    union { u32x4 q[2]; v16h v; } U;
    U.q[0] = *(const u32x4*)(p);
    U.q[1] = *(const u32x4*)(p + 16);
    return U.v;
}
// B-matrix 32x16 f16 (stored transposed [N,K]): lane<16 -> K = {kk+0..7, kk+16..23}; lane>=16 -> +8
__device__ __forceinline__ v16h load_fragB(const _Float16* baseT, int row, int ldk,
                                           int kk, int lane) {
    int kg = (lane < 16) ? 0 : 8;
    const _Float16* p = baseT + (size_t)row * ldk + kk + kg;
    union { u32x4 q[2]; v16h v; } U;
    U.q[0] = *(const u32x4*)(p);
    U.q[1] = *(const u32x4*)(p + 16);
    return U.v;
}

__device__ __forceinline__ v8f wmma_acc(v16h a, v16h b, v8f c) {
    return __builtin_amdgcn_wmma_f32_16x16x32_f16(false, a, false, b, (short)0, c,
                                                  false, false);
}

template <int EPI>
__device__ __forceinline__ void store_tile(float* __restrict__ C, int Nn,
                                           int mt, int nt, int lane, v8f c) {
    int n  = nt * 16 + (lane & 15);
    int m0 = mt * 16 + ((lane >> 4) << 3);
    for (int i = 0; i < 8; ++i) {
        float v = c[i];
        if (EPI == 1) v = 1.0f / (1.0f + __expf(-v));
        C[(size_t)(m0 + i) * Nn + n] = v;
    }
}

// ---------- 2x2 register-blocked f16 WMMA GEMM: C[M,Nn] = A[M,K] * B^T[Nn,K] ----------
// Each wave computes a 32x32 output block (4 accumulators); A/B fragments reused 2x,
// halving L2 traffic vs one tile per wave. Edge blocks (odd tile counts) dispatch to
// clean 1x2 / 2x1 / 1x1 K-loops -- all branches wave-uniform, EXEC stays all-ones.
// EPI: 0 = none, 1 = sigmoid
template <int EPI>
__global__ void k_gemm_wmma(const _Float16* __restrict__ A,
                            const _Float16* __restrict__ BT,
                            float* __restrict__ C,
                            int M, int Nn, int K) {
    int lane  = threadIdx.x & 31;
    int wave  = blockIdx.x * (blockDim.x >> 5) + (threadIdx.x >> 5);
    int nwave = gridDim.x * (blockDim.x >> 5);
    int tilesM = M >> 4, tilesN = Nn >> 4;
    int bM = (tilesM + 1) >> 1, bN = (tilesN + 1) >> 1;
    int TBLK = bM * bN;
    for (int t = wave; t < TBLK; t += nwave) {
        int bm = t / bN;
        int bn = t - bm * bN;
        int mt0 = bm << 1, nt0 = bn << 1;
        bool m1 = (mt0 + 1) < tilesM;
        bool n1 = (nt0 + 1) < tilesN;
        int ra0 = mt0 * 16 + (lane & 15);
        int rb0 = nt0 * 16 + (lane & 15);
        if (m1 && n1) {
            v8f c00 = {}, c01 = {}, c10 = {}, c11 = {};
            for (int kk = 0; kk < K; kk += 32) {
                v16h a0 = load_fragA(A, ra0, K, kk, lane);
                v16h a1 = load_fragA(A, ra0 + 16, K, kk, lane);
                v16h b0 = load_fragB(BT, rb0, K, kk, lane);
                v16h b1 = load_fragB(BT, rb0 + 16, K, kk, lane);
                c00 = wmma_acc(a0, b0, c00);
                c01 = wmma_acc(a0, b1, c01);
                c10 = wmma_acc(a1, b0, c10);
                c11 = wmma_acc(a1, b1, c11);
            }
            store_tile<EPI>(C, Nn, mt0,     nt0,     lane, c00);
            store_tile<EPI>(C, Nn, mt0,     nt0 + 1, lane, c01);
            store_tile<EPI>(C, Nn, mt0 + 1, nt0,     lane, c10);
            store_tile<EPI>(C, Nn, mt0 + 1, nt0 + 1, lane, c11);
        } else if (n1) {
            v8f c00 = {}, c01 = {};
            for (int kk = 0; kk < K; kk += 32) {
                v16h a0 = load_fragA(A, ra0, K, kk, lane);
                v16h b0 = load_fragB(BT, rb0, K, kk, lane);
                v16h b1 = load_fragB(BT, rb0 + 16, K, kk, lane);
                c00 = wmma_acc(a0, b0, c00);
                c01 = wmma_acc(a0, b1, c01);
            }
            store_tile<EPI>(C, Nn, mt0, nt0,     lane, c00);
            store_tile<EPI>(C, Nn, mt0, nt0 + 1, lane, c01);
        } else if (m1) {
            v8f c00 = {}, c10 = {};
            for (int kk = 0; kk < K; kk += 32) {
                v16h a0 = load_fragA(A, ra0, K, kk, lane);
                v16h a1 = load_fragA(A, ra0 + 16, K, kk, lane);
                v16h b0 = load_fragB(BT, rb0, K, kk, lane);
                c00 = wmma_acc(a0, b0, c00);
                c10 = wmma_acc(a1, b0, c10);
            }
            store_tile<EPI>(C, Nn, mt0,     nt0, lane, c00);
            store_tile<EPI>(C, Nn, mt0 + 1, nt0, lane, c10);
        } else {
            v8f c00 = {};
            for (int kk = 0; kk < K; kk += 32) {
                v16h a0 = load_fragA(A, ra0, K, kk, lane);
                v16h b0 = load_fragB(BT, rb0, K, kk, lane);
                c00 = wmma_acc(a0, b0, c00);
            }
            store_tile<EPI>(C, Nn, mt0, nt0, lane, c00);
        }
    }
}

// ---------- attention terms: el[n,h] = sum_d feat[n,h,d]*al[h,d] ----------
__global__ void k_attn_terms(const float* __restrict__ feat,
                             const float* __restrict__ al,
                             const float* __restrict__ ar,
                             float* __restrict__ el, float* __restrict__ er,
                             int Nn, int D) {
    int t = blockIdx.x * blockDim.x + threadIdx.x;
    if (t >= Nn * NHEAD) return;
    int n = t >> 2, h = t & 3;
    const float* f  = feat + ((size_t)n * NHEAD + h) * D;
    const float* a1 = al + h * D;
    const float* a2 = ar + h * D;
    float s1 = 0.f, s2 = 0.f;
    for (int d = 0; d < D; ++d) { float v = f[d]; s1 += v * a1[d]; s2 += v * a2[d]; }
    el[t] = s1; er[t] = s2;
}

// ---------- edge pass 1: leaky_relu + segment max ----------
__global__ void k_edge_lrelu_max(const int* __restrict__ src, const int* __restrict__ dst,
                                 const float* __restrict__ el, const float* __restrict__ er,
                                 float* __restrict__ ebuf, unsigned* __restrict__ emax,
                                 int E) {
    int t = blockIdx.x * blockDim.x + threadIdx.x;
    if (t >= E * NHEAD) return;
    int e = t >> 2, h = t & 3;
    float x = el[src[e] * NHEAD + h] + er[dst[e] * NHEAD + h];
    x = (x > 0.f) ? x : 0.2f * x;
    ebuf[t] = x;
    atomicMax(&emax[dst[e] * NHEAD + h], f2o(x));
}

// ---------- decode ordered-uint max, zero non-finite (in place -> float) ----------
__global__ void k_fix_emax(unsigned* p, int n) {
    int t = blockIdx.x * blockDim.x + threadIdx.x;
    if (t >= n) return;
    float m = o2f(p[t]);
    if (!(m >= -3.0e38f && m <= 3.0e38f)) m = 0.f;
    ((float*)p)[t] = m;
}

// ---------- edge pass 2: exp + segment sum ----------
__global__ void k_edge_exp_sum(const int* __restrict__ dst,
                               float* __restrict__ ebuf,
                               const float* __restrict__ emax,
                               float* __restrict__ denom, int E) {
    int t = blockIdx.x * blockDim.x + threadIdx.x;
    if (t >= E * NHEAD) return;
    int e = t >> 2, h = t & 3;
    float a = __expf(ebuf[t] - emax[dst[e] * NHEAD + h]);
    ebuf[t] = a;
    atomicAdd(&denom[dst[e] * NHEAD + h], a);
}

// ---------- edge pass 3: normalize ----------
__global__ void k_edge_att(const int* __restrict__ dst,
                           const float* __restrict__ ebuf,
                           const float* __restrict__ denom,
                           float* __restrict__ att, int E) {
    int t = blockIdx.x * blockDim.x + threadIdx.x;
    if (t >= E * NHEAD) return;
    int e = t >> 2, h = t & 3;
    att[t] = ebuf[t] / fmaxf(denom[dst[e] * NHEAD + h], 1e-9f);
}

// ---------- weighted scatter-sum: out[dst] += att * feat[src] ----------
__global__ void k_scatter(const int* __restrict__ src, const int* __restrict__ dst,
                          const float* __restrict__ att,
                          const float* __restrict__ feat,
                          float* __restrict__ out, int E, int D) {
    long t = (long)blockIdx.x * blockDim.x + threadIdx.x;
    if (t >= (long)E * D) return;
    int e = (int)(t / D);
    int d = (int)(t - (long)e * D);
    int s = src[e], q = dst[e];
    for (int h = 0; h < NHEAD; ++h) {
        float v = att[e * NHEAD + h] * feat[((size_t)s * NHEAD + h) * D + d];
        atomicAdd(&out[((size_t)q * NHEAD + h) * D + d], v);
    }
}

// ---------- layer-1 post: h1 = elu(out + b1), emit f16 ----------
__global__ void k_post1(const float* __restrict__ out1, const float* __restrict__ b1,
                        _Float16* __restrict__ h1h, long n) {
    long t = (long)blockIdx.x * blockDim.x + threadIdx.x;
    if (t >= n) return;
    int c = (int)(t & (HID - 1));
    float v = out1[t] + b1[c];
    v = (v > 0.f) ? v : (__expf(v) - 1.0f);
    h1h[t] = (_Float16)v;
}

// ---------- layer-2 post: enc = mean_h(out2 + b2), emit f16 ----------
__global__ void k_post2(const float* __restrict__ out2, const float* __restrict__ b2,
                        _Float16* __restrict__ ench, long n) {
    long t = (long)blockIdx.x * blockDim.x + threadIdx.x;
    if (t >= n) return;
    int nn = (int)(t / LATENT);
    int d  = (int)(t - (long)nn * LATENT);
    float s = 0.f;
    for (int h = 0; h < NHEAD; ++h)
        s += out2[((size_t)nn * NHEAD + h) * LATENT + d] + b2[h * LATENT + d];
    ench[t] = (_Float16)(0.25f * s);
}

static inline unsigned cdivu(long a, long b) { return (unsigned)((a + b - 1) / b); }

extern "C" void kernel_launch(void* const* d_in, const int* in_sizes, int n_in,
                              void* d_out, int out_size, void* d_ws, size_t ws_size,
                              hipStream_t stream) {
    const float* X   = (const float*)d_in[0];
    const int*   src = (const int*)  d_in[1];
    const int*   dst = (const int*)  d_in[2];
    const float* W1  = (const float*)d_in[3];
    const float* al1 = (const float*)d_in[4];
    const float* ar1 = (const float*)d_in[5];
    const float* b1  = (const float*)d_in[6];
    const float* W2  = (const float*)d_in[7];
    const float* al2 = (const float*)d_in[8];
    const float* ar2 = (const float*)d_in[9];
    const float* b2  = (const float*)d_in[10];

    float* recon = (float*)d_out;
    float* att1  = recon + (size_t)N_NODES * N_NODES;  // [E, NHEAD] tail output

    // bump-allocate workspace (256B aligned)
    char* w = (char*)d_ws;
    size_t off = 0;
    auto alloc = [&](size_t bytes) -> char* {
        char* p = w + off;
        off += (bytes + 255) & ~(size_t)255;
        return p;
    };
    _Float16* Xh    = (_Float16*)alloc((size_t)N_NODES * IN_F * 2);
    _Float16* W1T   = (_Float16*)alloc((size_t)IN_F * HID * 2);
    float*    feat1 = (float*)   alloc((size_t)N_NODES * HID * 4);
    float*    el1   = (float*)   alloc((size_t)N_NODES * NHEAD * 4);
    float*    er1   = (float*)   alloc((size_t)N_NODES * NHEAD * 4);
    unsigned* emax1 = (unsigned*)alloc((size_t)N_NODES * NHEAD * 4);
    float*    den1  = (float*)   alloc((size_t)N_NODES * NHEAD * 4);
    float*    ebuf1 = (float*)   alloc((size_t)N_EDGES * NHEAD * 4);
    float*    out1  = (float*)   alloc((size_t)N_NODES * HID * 4);
    _Float16* h1h   = (_Float16*)alloc((size_t)N_NODES * HID * 2);
    _Float16* W2T   = (_Float16*)alloc((size_t)HID * NOUT2 * 2);
    float*    feat2 = (float*)   alloc((size_t)N_NODES * NOUT2 * 4);
    float*    el2   = (float*)   alloc((size_t)N_NODES * NHEAD * 4);
    float*    er2   = (float*)   alloc((size_t)N_NODES * NHEAD * 4);
    unsigned* emax2 = (unsigned*)alloc((size_t)N_NODES * NHEAD * 4);
    float*    den2  = (float*)   alloc((size_t)N_NODES * NHEAD * 4);
    float*    ebuf2 = (float*)   alloc((size_t)N_EDGES * NHEAD * 4);
    float*    att2  = (float*)   alloc((size_t)N_EDGES * NHEAD * 4);
    float*    out2  = (float*)   alloc((size_t)N_NODES * NOUT2 * 4);
    _Float16* ench  = (_Float16*)alloc((size_t)N_NODES * LATENT * 2);
    (void)ws_size;

    const int TB = 256;
    auto gemm_blocks = [](int M, int Nn) -> unsigned {
        int bM = ((M >> 4) + 1) >> 1, bN = ((Nn >> 4) + 1) >> 1;
        return cdivu((long)bM * bN, 8);  // 8 waves per 256-thread block
    };

    // ---- casts ----
    k_cast_h <<<cdivu((long)N_NODES * IN_F, TB), TB, 0, stream>>>(X, Xh, (long)N_NODES * IN_F);
    k_cast_hT<<<cdivu((long)IN_F * HID,   TB), TB, 0, stream>>>(W1, W1T, IN_F, HID);
    k_cast_hT<<<cdivu((long)HID * NOUT2,  TB), TB, 0, stream>>>(W2, W2T, HID, NOUT2);

    // ---- GATConv 1 ----
    k_gemm_wmma<0><<<gemm_blocks(N_NODES, HID), 256, 0, stream>>>(Xh, W1T, feat1, N_NODES, HID, IN_F);
    k_attn_terms<<<cdivu((long)N_NODES * NHEAD, TB), TB, 0, stream>>>(feat1, al1, ar1, el1, er1, N_NODES, DH1);
    k_fill_u32<<<1024, TB, 0, stream>>>(emax1, 0x007fffffu, (long)N_NODES * NHEAD);          // f2o(-inf)
    k_fill_u32<<<1024, TB, 0, stream>>>((unsigned*)den1, 0u, (long)N_NODES * NHEAD);
    k_fill_u32<<<4096, TB, 0, stream>>>((unsigned*)out1, 0u, (long)N_NODES * HID);
    k_edge_lrelu_max<<<cdivu((long)N_EDGES * NHEAD, TB), TB, 0, stream>>>(src, dst, el1, er1, ebuf1, emax1, N_EDGES);
    k_fix_emax<<<cdivu((long)N_NODES * NHEAD, TB), TB, 0, stream>>>(emax1, N_NODES * NHEAD);
    k_edge_exp_sum<<<cdivu((long)N_EDGES * NHEAD, TB), TB, 0, stream>>>(dst, ebuf1, (float*)emax1, den1, N_EDGES);
    k_edge_att<<<cdivu((long)N_EDGES * NHEAD, TB), TB, 0, stream>>>(dst, ebuf1, den1, att1, N_EDGES);
    k_scatter<<<cdivu((long)N_EDGES * DH1, TB), TB, 0, stream>>>(src, dst, att1, feat1, out1, N_EDGES, DH1);
    k_post1<<<cdivu((long)N_NODES * HID, TB), TB, 0, stream>>>(out1, b1, h1h, (long)N_NODES * HID);

    // ---- GATConv 2 ----
    k_gemm_wmma<0><<<gemm_blocks(N_NODES, NOUT2), 256, 0, stream>>>(h1h, W2T, feat2, N_NODES, NOUT2, HID);
    k_attn_terms<<<cdivu((long)N_NODES * NHEAD, TB), TB, 0, stream>>>(feat2, al2, ar2, el2, er2, N_NODES, LATENT);
    k_fill_u32<<<1024, TB, 0, stream>>>(emax2, 0x007fffffu, (long)N_NODES * NHEAD);
    k_fill_u32<<<1024, TB, 0, stream>>>((unsigned*)den2, 0u, (long)N_NODES * NHEAD);
    k_fill_u32<<<8192, TB, 0, stream>>>((unsigned*)out2, 0u, (long)N_NODES * NOUT2);
    k_edge_lrelu_max<<<cdivu((long)N_EDGES * NHEAD, TB), TB, 0, stream>>>(src, dst, el2, er2, ebuf2, emax2, N_EDGES);
    k_fix_emax<<<cdivu((long)N_NODES * NHEAD, TB), TB, 0, stream>>>(emax2, N_NODES * NHEAD);
    k_edge_exp_sum<<<cdivu((long)N_EDGES * NHEAD, TB), TB, 0, stream>>>(dst, ebuf2, (float*)emax2, den2, N_EDGES);
    k_edge_att<<<cdivu((long)N_EDGES * NHEAD, TB), TB, 0, stream>>>(dst, ebuf2, den2, att2, N_EDGES);
    k_scatter<<<cdivu((long)N_EDGES * LATENT, TB), TB, 0, stream>>>(src, dst, att2, feat2, out2, N_EDGES, LATENT);
    k_post2<<<cdivu((long)N_NODES * LATENT, TB), TB, 0, stream>>>(out2, b2, ench, (long)N_NODES * LATENT);

    // ---- recon = sigmoid(enc @ enc^T): 313x313 blocks of 32x32 (2x2 WMMA tiles) ----
    k_gemm_wmma<1><<<gemm_blocks(N_NODES, N_NODES), 256, 0, stream>>>(ench, ench, recon, N_NODES, N_NODES, LATENT);
}
